// FourierCrossAttention_1726576856556
// MI455X (gfx1250) — matrix operand order repaired and checked by reading, not compiled, not run on hardware
//
#include <hip/hip_runtime.h>

// ---------------------------------------------------------------------------
// FourierCrossAttention, fully fused per (b,h) slice for MI455X (gfx1250).
//   - "rfft keep 64 modes"  == GEMM against 2048x128 cos/-sin basis (WMMA bf16)
//   - middle complex 64^3 contractions == 4 real WMMA GEMMs each
//   - per-mode weight contraction (w1) on VALU (tiny, L2-resident weights)
//   - irfft of 64 modes == GEMM against 128x2048 basis (WMMA bf16)
// Memory-bound: 768 MB mandatory HBM traffic -> ~33 us at 23.3 TB/s.
// ---------------------------------------------------------------------------

typedef __attribute__((ext_vector_type(16))) __bf16 v16bf;
typedef __attribute__((ext_vector_type(8)))  __bf16 v8bf;
typedef __attribute__((ext_vector_type(8)))  float  v8f;

#define B_  64
#define L_  2048
#define H_  8
#define E_  64
#define M_  64      // modes kept
#define O_  64

static constexpr int PIT = 72;            // LDS pitch (bf16) for 64-wide arrays (+8 pad, keeps 16B align)
static constexpr int YP  = 136;           // LDS pitch for Y (K=128, +8 pad)
static constexpr int A64 = 64 * PIT * 2;  // bytes of one 64x64 bf16 LDS array = 9216

// Phase-overlaid LDS layout (bytes)
static constexpr int OF_XQT_RE = 0;        // Xq^T [x][e]      (phase1 -> 2a)
static constexpr int OF_XQT_IM = A64;
static constexpr int OF_XKT_RE = 2 * A64;  // Xk^T [y][e]      (phase1 -> 2a)
static constexpr int OF_XKT_IM = 3 * A64;
static constexpr int OF_XKC_RE = 4 * A64;  // Xk   [e][y]      (phase1 -> 2b)
static constexpr int OF_XKC_IM = 5 * A64;
static constexpr int OF_STG    = 6 * A64;  // phase-1 staging tile [e][t]
static constexpr int OF_XQK_RE = 6 * A64;  // tanh(Xq^T Xk) [x][y]  (aliases STG, dead after phase1)
static constexpr int OF_XQK_IM = 7 * A64;
static constexpr int OF_XQV_RE = 0;        // xqkv [x][e]      (aliases XQT, dead after 2a)
static constexpr int OF_XQV_IM = A64;
static constexpr int OF_Y      = 2 * A64;  // Y [o][128]       (aliases XKT, dead after 2a) 64*136*2=17408<=2*A64
static constexpr int SMEM_BYTES = 8 * A64; // 73728 (< 320KB gfx1250 WG limit)

__device__ __forceinline__ v8f vz() {
  v8f z = {0.f, 0.f, 0.f, 0.f, 0.f, 0.f, 0.f, 0.f};
  return z;
}

__device__ __forceinline__ v8f wmma_bf16(v16bf a, v16bf b, v8f c) {
  // (neg_a, A, neg_b, B, c_mod, C, reuse_a, reuse_b)
  return __builtin_amdgcn_wmma_f32_16x16x32_bf16(false, a, false, b, (short)0, c, false, false);
}

// A fragment (16x32, 16-bit): lanes 0-15 row M, K {0..7,16..23}; lanes 16-31 K {8..15,24..31}
// Source matrix is row-major [M][K] with pitch `pitch` (bf16 elements, multiple of 8).
__device__ __forceinline__ v16bf ldA(const __bf16* __restrict__ base, int pitch, int row0, int kb) {
  const int lane = threadIdx.x & 31;
  const __bf16* p = base + (size_t)(row0 + (lane & 15)) * pitch + kb + ((lane & 16) ? 8 : 0);
  v8bf lo = *(const v8bf*)p;          // K pattern +0..7
  v8bf hi = *(const v8bf*)(p + 16);   // K pattern +16..23
  return __builtin_shufflevector(lo, hi, 0,1,2,3,4,5,6,7,8,9,10,11,12,13,14,15);
}

// B fragment (32x16, 16-bit): lanes 0-15 col N, K {0..7,16..23}; lanes 16-31 K {8..15,24..31}.
// Source is B^T row-major [N][K] with pitch `pitch`.
__device__ __forceinline__ v16bf ldB(const __bf16* __restrict__ base, int pitch, int row0, int kb) {
  const int lane = threadIdx.x & 31;
  const __bf16* p = base + (size_t)(row0 + (lane & 15)) * pitch + kb + ((lane & 16) ? 8 : 0);
  v8bf lo = *(const v8bf*)p;
  v8bf hi = *(const v8bf*)(p + 16);
  return __builtin_shufflevector(lo, hi, 0,1,2,3,4,5,6,7,8,9,10,11,12,13,14,15);
}

// ------------------------- basis generation kernel -------------------------
__global__ void fca_basis(__bf16* __restrict__ WfT, __bf16* __restrict__ WiT) {
  const int idx = blockIdx.x * 256 + threadIdx.x;     // 0 .. 262143
  const double PI1024 = 3.14159265358979323846 / 1024.0;
  // Forward basis, B^T layout [n=0..127][t=0..2047]: n<64 -> cos, n>=64 -> -sin
  {
    int n = idx >> 11, t = idx & 2047;
    int mode = (n < 64) ? n : (n - 64);
    int r = (mode * t) & 2047;                        // exact periodic reduction
    double a = (double)r * PI1024;
    double v = (n < 64) ? cos(a) : -sin(a);
    WfT[idx] = (__bf16)(float)v;
  }
  // Inverse basis, B^T layout [t=0..2047][k=0..127]; scale = 1/(L*IN_CH*OUT_CH),
  // Hermitian factor 2 for modes>=1, Im of mode 0 dropped (irfft semantics).
  {
    int t = idx >> 7, kk = idx & 127;
    int mode = (kk < 64) ? kk : (kk - 64);
    int r = (mode * t) & 2047;
    double a = (double)r * PI1024;
    double S = 1.0 / (2048.0 * 512.0 * 512.0);
    double alpha = (mode == 0) ? ((kk < 64) ? 1.0 : 0.0) : 2.0;
    double v = (kk < 64) ? (alpha * cos(a) * S) : (-alpha * sin(a) * S);
    WiT[idx] = (__bf16)(float)v;
  }
}

// ------------------------- fused main kernel -------------------------------
// Phase 1: DFT of one (b,h) slice: out[e, col] = sum_t x[t,e] * Wf[t, col], col=0..127
// writes transposed spectra (and natural copy for k) into LDS as bf16.
__device__ void dft_phase(const float* __restrict__ src_bh,   // &x[b,0,h,0]
                          const __bf16* __restrict__ WfT,
                          __bf16* __restrict__ stg,
                          __bf16* __restrict__ T_re, __bf16* __restrict__ T_im,
                          __bf16* __restrict__ C_re, __bf16* __restrict__ C_im) {
  const int tid = threadIdx.x;
  const int w = tid >> 5, lane = tid & 31;
  const int mt = w & 3, nb = (w >> 2) * 4;     // wave: M-tile mt, N-tiles nb..nb+3
  const int hi = lane >> 4, r = lane & 15;
  v8f acc[4] = {vz(), vz(), vz(), vz()};

  for (int kt0 = 0; kt0 < L_; kt0 += 64) {
    __syncthreads();                            // staging buffer free to overwrite
    #pragma unroll
    for (int it = 0; it < 16; ++it) {           // 64t x 64e tile, coalesced over e
      int idx = it * 256 + tid;
      int tt = idx >> 6, e = idx & 63;
      float v = src_bh[(size_t)(kt0 + tt) * (H_ * E_) + e];
      stg[e * PIT + tt] = (__bf16)v;            // transpose: rows = e (WMMA M)
    }
    __syncthreads();
    #pragma unroll
    for (int ks = 0; ks < 64; ks += 32) {
      v16bf a = ldA(stg, PIT, mt * 16, ks);
      #pragma unroll
      for (int j = 0; j < 4; ++j) {
        v16bf bf = ldB(WfT, 2048, (nb + j) * 16, kt0 + ks);   // basis: L2-hot
        acc[j] = wmma_bf16(a, bf, acc[j]);
      }
    }
  }
  __syncthreads();
  // scatter C tiles: col<64 -> Re(mode col), col>=64 -> Im(mode col-64)
  #pragma unroll
  for (int j = 0; j < 4; ++j) {
    int col = (nb + j) * 16 + r;
    #pragma unroll
    for (int i = 0; i < 8; ++i) {
      int m = mt * 16 + i + 8 * hi;             // e index
      __bf16 bv = (__bf16)acc[j][i];
      if (col < 64) {
        T_re[col * PIT + m] = bv;               // transposed [mode][e]
        if (C_re) C_re[m * PIT + col] = bv;     // natural   [e][mode]
      } else {
        int cm = col - 64;
        T_im[cm * PIT + m] = bv;
        if (C_im) C_im[m * PIT + cm] = bv;
      }
    }
  }
}

__global__ __launch_bounds__(256)
void fca_main(const float* __restrict__ q, const float* __restrict__ k,
              const float* __restrict__ w1r, const float* __restrict__ w1i,
              const __bf16* __restrict__ WfT, const __bf16* __restrict__ WiT,
              float* __restrict__ out) {
  __shared__ __align__(16) char smem[SMEM_BYTES];
  const int bh = blockIdx.x;
  const int b = bh >> 3, h = bh & 7;
  const int tid = threadIdx.x;
  const int w = tid >> 5, lane = tid & 31;
  const int hi = lane >> 4, r = lane & 15;

  __bf16* XQT_re = (__bf16*)(smem + OF_XQT_RE);
  __bf16* XQT_im = (__bf16*)(smem + OF_XQT_IM);
  __bf16* XKT_re = (__bf16*)(smem + OF_XKT_RE);
  __bf16* XKT_im = (__bf16*)(smem + OF_XKT_IM);
  __bf16* XKC_re = (__bf16*)(smem + OF_XKC_RE);
  __bf16* XKC_im = (__bf16*)(smem + OF_XKC_IM);
  __bf16* STG    = (__bf16*)(smem + OF_STG);
  __bf16* XQK_re = (__bf16*)(smem + OF_XQK_RE);
  __bf16* XQK_im = (__bf16*)(smem + OF_XQK_IM);
  __bf16* XQV_re = (__bf16*)(smem + OF_XQV_RE);
  __bf16* XQV_im = (__bf16*)(smem + OF_XQV_IM);
  __bf16* Yl     = (__bf16*)(smem + OF_Y);

  // ---- Phase 1: 64-mode DFT of q and k slices -----------------------------
  const float* qbh = q + ((size_t)b * L_ * H_ + h) * E_;
  const float* kbh = k + ((size_t)b * L_ * H_ + h) * E_;
  dft_phase(qbh, WfT, STG, XQT_re, XQT_im, nullptr, nullptr);
  dft_phase(kbh, WfT, STG, XKT_re, XKT_im, XKC_re, XKC_im);
  __syncthreads();

  // ---- Phase 2a: XQK[x,y] = tanh( sum_e Xq[e,x]*Xk[e,y] )  (complex) ------
  #pragma unroll
  for (int tt = 2 * w; tt <= 2 * w + 1; ++tt) {
    int mt = tt >> 2, nt = tt & 3;
    v8f crr = vz(), cii = vz(), cim = vz();
    #pragma unroll
    for (int kb = 0; kb < 64; kb += 32) {
      v16bf are = ldA(XQT_re, PIT, mt * 16, kb);
      v16bf aim = ldA(XQT_im, PIT, mt * 16, kb);
      v16bf bre = ldB(XKT_re, PIT, nt * 16, kb);
      v16bf bim = ldB(XKT_im, PIT, nt * 16, kb);
      crr = wmma_bf16(are, bre, crr);
      cii = wmma_bf16(aim, bim, cii);
      cim = wmma_bf16(are, bim, cim);
      cim = wmma_bf16(aim, bre, cim);
    }
    #pragma unroll
    for (int i = 0; i < 8; ++i) {
      float a  = crr[i] - cii[i];               // Re = rr - ii
      float bb = cim[i];                        // Im = ri + ir
      // complex tanh: (sinh 2a + i sin 2b) / (cosh 2a + cos 2b)
      float t2a = fminf(fmaxf(2.f * a, -40.f), 40.f);
      float ea = __expf(t2a), eai = __expf(-t2a);
      float sh = 0.5f * (ea - eai), ch = 0.5f * (ea + eai);
      float s, c;
      __sincosf(2.f * bb, &s, &c);
      float inv = 1.0f / (ch + c);
      int m = mt * 16 + i + 8 * hi, n = nt * 16 + r;
      XQK_re[m * PIT + n] = (__bf16)(sh * inv);
      XQK_im[m * PIT + n] = (__bf16)(s * inv);
    }
  }
  __syncthreads();

  // ---- Phase 2b: xqkv[x,e] = sum_y XQK[x,y] * Xk[e,y]  (complex) ----------
  #pragma unroll
  for (int tt = 2 * w; tt <= 2 * w + 1; ++tt) {
    int mt = tt >> 2, nt = tt & 3;              // mt: x-tile, nt: e-tile
    v8f crr = vz(), cii = vz(), cim = vz();
    #pragma unroll
    for (int kb = 0; kb < 64; kb += 32) {       // kb over y
      v16bf are = ldA(XQK_re, PIT, mt * 16, kb);
      v16bf aim = ldA(XQK_im, PIT, mt * 16, kb);
      v16bf bre = ldB(XKC_re, PIT, nt * 16, kb);
      v16bf bim = ldB(XKC_im, PIT, nt * 16, kb);
      crr = wmma_bf16(are, bre, crr);
      cii = wmma_bf16(aim, bim, cii);
      cim = wmma_bf16(are, bim, cim);
      cim = wmma_bf16(aim, bre, cim);
    }
    #pragma unroll
    for (int i = 0; i < 8; ++i) {
      int m = mt * 16 + i + 8 * hi, n = nt * 16 + r;   // m: x, n: e
      XQV_re[m * PIT + n] = (__bf16)(crr[i] - cii[i]);
      XQV_im[m * PIT + n] = (__bf16)cim[i];
    }
  }
  __syncthreads();

  // ---- Phase 2c: Y[o,x] = sum_e xqkv[x,e] * w1[h,e,o,x]  (complex, VALU) --
  {
    const int x = tid & 63, o0 = (tid >> 6) * 16;     // lanes sweep x -> coalesced w1
    float yr[16], yi[16];
    #pragma unroll
    for (int oi = 0; oi < 16; ++oi) { yr[oi] = 0.f; yi[oi] = 0.f; }
    for (int e = 0; e < 64; ++e) {
      float xr = (float)XQV_re[x * PIT + e];
      float xi = (float)XQV_im[x * PIT + e];
      const float* wr = w1r + (((size_t)h * E_ + e) * O_ + o0) * M_ + x;
      const float* wi = w1i + (((size_t)h * E_ + e) * O_ + o0) * M_ + x;
      #pragma unroll
      for (int oi = 0; oi < 16; ++oi) {
        float a = wr[(size_t)oi * M_], bq = wi[(size_t)oi * M_];
        yr[oi] += xr * a - xi * bq;
        yi[oi] += xr * bq + xi * a;
      }
    }
    #pragma unroll
    for (int oi = 0; oi < 16; ++oi) {
      Yl[(o0 + oi) * YP + x]      = (__bf16)yr[oi];   // K cols 0..63  = Re
      Yl[(o0 + oi) * YP + 64 + x] = (__bf16)yi[oi];   // K cols 64..127 = Im
    }
  }
  __syncthreads();

  // ---- Phase 3: iDFT  out[o,t] = sum_k Y[o,k] * Wi[k,t]  (M=64,N=2048,K=128)
  {
    v16bf aY[4][4];
    #pragma unroll
    for (int mt = 0; mt < 4; ++mt)
      #pragma unroll
      for (int kk = 0; kk < 4; ++kk)
        aY[mt][kk] = ldA(Yl, YP, mt * 16, kk * 32);

    const size_t obase = (size_t)bh * O_ * L_;
    for (int ntl = 0; ntl < 16; ++ntl) {              // each wave: 256 t values
      int t0 = (w * 16 + ntl) * 16;
      v8f c[4] = {vz(), vz(), vz(), vz()};
      #pragma unroll
      for (int kk = 0; kk < 4; ++kk) {
        v16bf bf = ldB(WiT, 128, t0, kk * 32);        // basis: L2-hot
        #pragma unroll
        for (int mt = 0; mt < 4; ++mt)
          c[mt] = wmma_bf16(aY[mt][kk], bf, c[mt]);
      }
      #pragma unroll
      for (int mt = 0; mt < 4; ++mt)
        #pragma unroll
        for (int i = 0; i < 8; ++i)
          out[obase + (size_t)(mt * 16 + i + 8 * hi) * L_ + t0 + r] = c[mt][i];
    }
  }
}

// --------------------------------------------------------------------------
extern "C" void kernel_launch(void* const* d_in, const int* in_sizes, int n_in,
                              void* d_out, int out_size, void* d_ws, size_t ws_size,
                              hipStream_t stream) {
  const float* q   = (const float*)d_in[0];
  const float* k   = (const float*)d_in[1];
  // d_in[2] = v (unused by the reference), d_in[5] = mask (unused)
  const float* w1r = (const float*)d_in[3];
  const float* w1i = (const float*)d_in[4];

  __bf16* WfT = (__bf16*)d_ws;                              // 128*2048 bf16 = 512 KB
  __bf16* WiT = (__bf16*)((char*)d_ws + 128 * 2048 * 2);    // 2048*128 bf16 = 512 KB

  fca_basis<<<1024, 256, 0, stream>>>(WfT, WiT);
  fca_main<<<B_ * H_, 256, 0, stream>>>(q, k, w1r, w1i, WfT, WiT, (float*)d_out);
}
